// NAIS_region_distance_disentangled_Embedding_77395310674425
// MI455X (gfx1250) — matrix operand order repaired and checked by reading, not compiled, not run on hardware
//
#include <hip/hip_runtime.h>
#include <math.h>

typedef __attribute__((ext_vector_type(16))) __bf16 v16bf;
typedef __attribute__((ext_vector_type(8)))  float  v8f;

#define B_   1024
#define H_   200
#define D_   128
#define MT   13      // ceil(200/16) M-tiles per batch

// f32 -> bf16 round-to-nearest-even, raw bits
__device__ __forceinline__ unsigned short f2bf(float f) {
    union { float f; unsigned u; } v; v.f = f;
    unsigned u = v.u;
    u += 0x7FFFu + ((u >> 16) & 1u);
    return (unsigned short)(u >> 16);
}

// ---------------------------------------------------------------------------
// Prep: pack W1 / Wr1 (f32 [128,128], row-major over d) into WMMA *A*-fragment
// order (bf16).  A frag (16x32, 16-bit): for element (n_local, kk):
//   lane = n_local + 16*((kk>>3)&1),  e = (kk&7) + 8*(kk>>4)
// Apack index: (((mat*8 + ntile)*4 + c)*32 + lane)*16 + e
// ---------------------------------------------------------------------------
__global__ void __launch_bounds__(32)
pack_w_kernel(const float* __restrict__ W1, const float* __restrict__ Wr1,
              unsigned short* __restrict__ Apack) {
    int bx   = blockIdx.x;            // 0..63 : mat*32 + nt*4 + c
    int lane = threadIdx.x;           // 0..31
    int c    = bx & 3;
    int nt   = (bx >> 2) & 7;
    int mat  = bx >> 5;
    const float* W = mat ? Wr1 : W1;
    int nl   = lane & 15;
    int hi   = lane >> 4;
    int base = (bx * 32 + lane) * 16;
#pragma unroll
    for (int e = 0; e < 16; ++e) {
        int kk = (e & 7) | (hi << 3) | ((e >> 3) << 4);
        Apack[base + e] = f2bf(W[(nt * 16 + nl) * D_ + c * 32 + kk]);
    }
}

// 128-thread block sum
__device__ __forceinline__ float blk_sum(float v, float* sh) {
    int tid = threadIdx.x;
    sh[tid] = v;
    __syncthreads();
    for (int s = 64; s > 0; s >>= 1) {
        if (tid < s) sh[tid] += sh[tid + s];
        __syncthreads();
    }
    float r = sh[0];
    __syncthreads();
    return r;
}

// ---------------------------------------------------------------------------
// Fused kernel: one 128-thread (4-wave) block per batch.
//   loop over 13 M-tiles: gather rows -> bf16 B frags -> WMMA (W = A operand)
//   -> bias/ReLU/w2 -> in-lane n-reduction; then masked-exp normalization and
//   sigmoid prediction, all in LDS.
// ---------------------------------------------------------------------------
__global__ void __launch_bounds__(128)
nais_fused_kernel(const int*   __restrict__ history,
                  const int*   __restrict__ target,
                  const int*   __restrict__ history_region,
                  const int*   __restrict__ target_region,
                  const float* __restrict__ target_distance,
                  const float* __restrict__ embed_history,
                  const float* __restrict__ embed_target,
                  const float* __restrict__ embed_region,
                  const float* __restrict__ embed_distance,
                  const unsigned short* __restrict__ Apack,
                  const float* __restrict__ b1,  const float* __restrict__ w2,
                  const float* __restrict__ br1, const float* __restrict__ wr2,
                  float* __restrict__ out) {
    __shared__ __align__(16) unsigned short ldsW[2 * 8 * 4 * 32 * 16]; // 64 KB
    __shared__ __align__(16) unsigned short ldsB[2 * 4 * 32 * 16];     //  8 KB
    __shared__ float tl[D_], trl[D_];
    __shared__ float red[2][16][8];
    __shared__ float apart[4][16];
    __shared__ float logitS[2][MT * 16];
    __shared__ float dotS[2][MT * 16];
    __shared__ float sh[128];

    const int b   = blockIdx.x;
    const int tid = threadIdx.x;

    // ---- copy packed W fragments into LDS (once per block) ----
    {
        const uint4* src = (const uint4*)Apack;
        uint4*       dst = (uint4*)ldsW;
#pragma unroll 4
        for (int i = tid; i < 4096; i += 128) dst[i] = src[i];
    }
    // ---- scalar: sum of embed_distance row 0 ----
    float se;
    {
        sh[tid] = embed_distance[tid];
        __syncthreads();
        for (int s = 64; s > 0; s >>= 1) {
            if (tid < s) sh[tid] += sh[tid + s];
            __syncthreads();
        }
        se = sh[0];
        __syncthreads();
    }
    // ---- target embeddings ----
    {
        int tg  = target[b];
        int trg = target_region[b];
        tl[tid]  = embed_target[(long)tg  * D_ + tid];
        trl[tid] = embed_region[(long)trg * D_ + tid];
    }
    __syncthreads();

    const int wave = tid >> 5, lane = tid & 31;
    const int r = tid >> 3, dseg = tid & 7;

    for (int mtile = 0; mtile < MT; ++mtile) {
        // ---- stage 2: gather 16 rows, build bf16 B fragments + f32 dots ----
        {
            int hg   = mtile * 16 + r;
            int hc   = hg < H_ ? hg : (H_ - 1);
            int hidx = history[b * H_ + hc];
            int ridx = history_region[b * H_ + hc];
            const float* hrow = embed_history + (long)hidx * D_;
            const float* rrow = embed_region  + (long)ridx * D_;
            float sA = 0.f, sR = 0.f;
#pragma unroll
            for (int j = 0; j < 16; ++j) {
                int d = dseg * 16 + j;
                float p  = hrow[d] * tl[d];
                float pr = rrow[d] * trl[d];
                sA += p; sR += pr;
                // B frag (32x16): lane = r+16*((kk>>3)&1), e = (kk&7)+8*(kk>>4)
                int c = d >> 5, kk = d & 31;
                int bl  = r + 16 * ((kk >> 3) & 1);
                int idx = (c * 32 + bl) * 16 + ((kk & 7) + 8 * (kk >> 4));
                ldsB[idx]        = f2bf(p);
                ldsB[2048 + idx] = f2bf(pr);
            }
            red[0][r][dseg] = sA;
            red[1][r][dseg] = sR;
        }
        __syncthreads();

        // ---- stage 3: WMMA.  waves 0,1 -> mat 0; waves 2,3 -> mat 1 ----
        {
            int mat = wave >> 1;
            int hi  = lane >> 4;
            const float* bp1 = mat ? br1 : b1;
            const float* wp1 = mat ? wr2 : w2;
            float am = 0.f;                       // per-lane: fixed m = lane&15
#pragma unroll
            for (int t = 0; t < 4; ++t) {
                int nt = (wave & 1) * 4 + t;
                v8f acc = {};
                const v16bf* ap  = (const v16bf*)&ldsW[(mat * 8 + nt) * 2048];
                const v16bf* bpv = (const v16bf*)&ldsB[mat * 2048];
#pragma unroll
                for (int c = 0; c < 4; ++c) {
                    acc = __builtin_amdgcn_wmma_f32_16x16x32_bf16(
                            false, ap[c * 32 + lane], false, bpv[c * 32 + lane],
                            (short)0, acc, false, false);
                }
                int nb = nt * 16 + 8 * hi;        // n = nb + i
#pragma unroll
                for (int i = 0; i < 8; ++i) {
                    float v = acc[i] + bp1[nb + i];
                    v = fmaxf(v, 0.f) * wp1[nb + i];
                    am += v;                      // in-lane sum over n
                }
            }
            am += __shfl_xor(am, 16, 32);         // combine n half-groups
            if (lane < 16) apart[wave][lane] = am;
        }
        __syncthreads();

        // ---- stage 4: fold wave partials, stash per-h logit + dot in LDS ----
        if (tid < 32) {
            int mat = tid >> 4, m = tid & 15;
            int hg = mtile * 16 + m;
            if (hg < H_) {
                float lg = apart[2 * mat][m] + apart[2 * mat + 1][m];
                float s = 0.f;
#pragma unroll
                for (int j = 0; j < 8; ++j) s += red[mat][m][j];
                logitS[mat][hg] = lg;
                dotS[mat][hg]   = s;
            }
        }
        __syncthreads();
    }

    // ---- final: masked exp, beta=0.5 norm, prediction, sigmoid ----
    int tgt = target[b];
    float eA0 = 0.f, eR0 = 0.f, dA0 = 0.f, dR0 = 0.f;
    float eA1 = 0.f, eR1 = 0.f, dA1 = 0.f, dR1 = 0.f;
    {
        int h = tid;                              // always < 200
        float dist = target_distance[b * H_ + h] * se;
        float msk  = (history[b * H_ + h] != tgt) ? 1.f : 0.f;
        eA0 = __expf(logitS[0][h] + dist) * msk;
        eR0 = __expf(logitS[1][h] + dist) * msk;
        dA0 = dotS[0][h]; dR0 = dotS[1][h];
        h = tid + 128;
        if (h < H_) {
            dist = target_distance[b * H_ + h] * se;
            msk  = (history[b * H_ + h] != tgt) ? 1.f : 0.f;
            eA1 = __expf(logitS[0][h] + dist) * msk;
            eR1 = __expf(logitS[1][h] + dist) * msk;
            dA1 = dotS[0][h]; dR1 = dotS[1][h];
        }
    }
    float sumA = blk_sum(eA0 + eA1, sh);
    float sumR = blk_sum(eR0 + eR1, sh);
    float nA = sqrtf(sumA);                       // BETA = 0.5
    float nR = sqrtf(sumR);
    float contrib = (eA0 * dA0 + eA1 * dA1) / nA + (eR0 * dR0 + eR1 * dR1) / nR;
    float pred = blk_sum(contrib, sh);
    if (tid == 0) out[b] = 1.f / (1.f + __expf(-pred));
}

// ---------------------------------------------------------------------------
extern "C" void kernel_launch(void* const* d_in, const int* in_sizes, int n_in,
                              void* d_out, int out_size, void* d_ws, size_t ws_size,
                              hipStream_t stream) {
    const int*   history         = (const int*)  d_in[0];
    const int*   target          = (const int*)  d_in[1];
    const int*   history_region  = (const int*)  d_in[2];
    const int*   target_region   = (const int*)  d_in[3];
    const float* target_distance = (const float*)d_in[4];
    const float* embed_history   = (const float*)d_in[5];
    const float* embed_target    = (const float*)d_in[6];
    const float* embed_region    = (const float*)d_in[7];
    const float* embed_distance  = (const float*)d_in[8];
    const float* W1  = (const float*)d_in[9];
    const float* b1  = (const float*)d_in[10];
    const float* w2  = (const float*)d_in[11];
    const float* Wr1 = (const float*)d_in[12];
    const float* br1 = (const float*)d_in[13];
    const float* wr2 = (const float*)d_in[14];
    float* out = (float*)d_out;

    unsigned short* Apack = (unsigned short*)d_ws;   // 64 KB packed weights

    pack_w_kernel<<<64, 32, 0, stream>>>(W1, Wr1, Apack);
    nais_fused_kernel<<<B_, 128, 0, stream>>>(
        history, target, history_region, target_region, target_distance,
        embed_history, embed_target, embed_region, embed_distance,
        Apack, b1, w2, br1, wr2, out);
}
